// Network_1288490189207
// MI455X (gfx1250) — matrix operand order repaired and checked by reading, not compiled, not run on hardware
//
#include <hip/hip_runtime.h>
#include <hip/hip_bf16.h>

// ---------------------------------------------------------------------------
// CDNA5 WMMA helpers (wave32, v_wmma_f32_16x16x32_f16)
// ---------------------------------------------------------------------------
typedef __attribute__((ext_vector_type(16))) _Float16 v16h;
typedef __attribute__((ext_vector_type(8)))  float    v8f;
typedef __attribute__((ext_vector_type(4)))  float    v4f;
typedef __attribute__((ext_vector_type(4)))  unsigned int v4u;
typedef __attribute__((ext_vector_type(8)))  int      v8i;
typedef __attribute__((ext_vector_type(4)))  int      v4i;

static __device__ __forceinline__ int lane_id() { return threadIdx.x & 31; }

// A-fragment (16x32 f16, row-major source, ld in halfs, ld % 8 == 0).
// Layout (ISA 7.12.2): lanes 0-15 row M=lane, K=[0..7] in v0..3, K=[16..23] in v4..7;
// lanes 16-31 row M=lane-16, K=[8..15] and K=[24..31].
static __device__ __forceinline__ v16h load_a_frag(const _Float16* A, int row_base,
                                                   int ld, int k0) {
  const int lane = lane_id();
  const int r    = lane & 15;
  const int kofs = (lane < 16) ? 0 : 8;
  const _Float16* p = A + (size_t)(row_base + r) * ld + (k0 + kofs);
  v16h out;
  ((v4f*)&out)[0] = *(const v4f*)(p);        // 8 halves: K = k0+kofs .. +7
  ((v4f*)&out)[1] = *(const v4f*)(p + 16);   // 8 halves: K = k0+16+kofs .. +7
  return out;
}

// B-fragment pre-packed: frag*512 + lane*16 + j  (j = K offset within half-tile)
static __device__ __forceinline__ v16h load_b_frag(const _Float16* B, int frag) {
  const int lane = lane_id();
  const _Float16* p = B + (size_t)frag * 512 + lane * 16;
  v16h out;
  ((v4f*)&out)[0] = *(const v4f*)(p);
  ((v4f*)&out)[1] = *(const v4f*)(p + 8);
  return out;
}

static __device__ __forceinline__ v8f wmma16(v16h a, v16h b, v8f c) {
  // (neg_a, A, neg_b, B, c_mod, C, reuse_a, reuse_b)
  return __builtin_amdgcn_wmma_f32_16x16x32_f16(false, a, false, b, (short)0, c,
                                                false, false);
}

// TDM: stage contiguous global f16 data into LDS via tensor_load_to_lds.
// Issued by one wave; caller must barrier before cross-wave use.
static __device__ __forceinline__ void tdm_stage_1d(const _Float16* gsrc,
                                                    _Float16* lds_dst,
                                                    unsigned elems /* f16 count */) {
#if __has_builtin(__builtin_amdgcn_tensor_load_to_lds) && \
    __has_builtin(__builtin_amdgcn_s_wait_tensorcnt)
  unsigned long long ga = (unsigned long long)gsrc;
  unsigned la = (unsigned)(unsigned long long)lds_dst;  // LDS offset = addr[31:0]
  // D# group0: count=1 | lds_addr | global_addr[56:0] | type=2
  v4u g0 = {1u, la, (unsigned)ga,
            (unsigned)((ga >> 32) & 0x1FFFFFFu) | (2u << 30)};
  // D# group1: data_size=1 (2B); tensor_dim0=elems @bits[79:48];
  // tensor_dim1=1 @bits[111:80]; tile_dim0=elems @bits[127:112]; stride0=elems.
  v8i g1 = {(int)(1u << 16),
            (int)((elems & 0xFFFFu) << 16),
            (int)((elems >> 16) | (1u << 16)),
            (int)(elems << 16),
            0,
            (int)elems,
            0, 0};
  v4i gz = {0, 0, 0, 0};
#if __clang_major__ >= 23
  v8i gz8 = {0, 0, 0, 0, 0, 0, 0, 0};
  __builtin_amdgcn_tensor_load_to_lds(g0, g1, gz, gz, gz8, 0);
#else
  __builtin_amdgcn_tensor_load_to_lds(g0, g1, gz, gz, 0);
#endif
  __builtin_amdgcn_s_wait_tensorcnt(0);
#else
  // Fallback: simple wave-cooperative copy (lane-strided b128s)
  const int lane = lane_id();
  for (unsigned i = lane * 8; i < elems; i += 32 * 8)
    *(v4f*)(lds_dst + i) = *(const v4f*)(gsrc + i);
#endif
}

// ---------------------------------------------------------------------------
// Weight repack: row-major f32 (optionally transposed, i.e. B[k][n]=w[n][k])
// -> f16 B-fragment layout, with scale folded in, zero-padded to tiles.
// ---------------------------------------------------------------------------
__global__ void pack_b_kernel(const float* __restrict__ w, _Float16* __restrict__ dst,
                              int ktiles, int ntiles, int Kreal, int Nreal,
                              int ldw, int transpose, float scale) {
  int idx = blockIdx.x * blockDim.x + threadIdx.x;
  int total = ktiles * ntiles * 512;
  if (idx >= total) return;
  int frag = idx >> 9;
  int lane = (idx >> 4) & 31;
  int j    = idx & 15;
  int kt = frag / ntiles, nt = frag - kt * ntiles;
  int n = nt * 16 + (lane & 15);
  int k = kt * 32 + ((lane < 16) ? 0 : 8) + (j & 7) + ((j & 8) << 1);
  float v = 0.f;
  if (k < Kreal && n < Nreal)
    v = transpose ? w[(size_t)n * ldw + k] : w[(size_t)k * ldw + n];
  dst[idx] = (_Float16)(v * scale);
}

// ---------------------------------------------------------------------------
// Generic WMMA GEMM: C[rows x ntiles*16] = A(f16,row-major) @ Bfrag (+ addend)
// One wave per 16-row tile. Compile-time KT (<=3) and HAS_ADD for branch-free
// straight-line WMMA chains. Epilogue is three independent unrolled passes
// (batch-load addend / add / batch-store) so the loads form one clause with a
// single wait instead of 8 serialized load-add-store round trips.
// ---------------------------------------------------------------------------
template <int KT, bool HAS_ADD>
__global__ void __launch_bounds__(256)
gemm_wmma_kernel(const _Float16* __restrict__ A, int lda,
                 const _Float16* __restrict__ B,
                 int rowTiles, int ntiles,
                 float* __restrict__ C, int ldc,
                 const float* __restrict__ add, int ldadd) {
  const int wv = blockIdx.x * (blockDim.x >> 5) + (threadIdx.x >> 5);
  if (wv >= rowTiles) return;
  const int rb   = wv * 16;
  const int lane = threadIdx.x & 31;
  const int cr   = lane & 15;
  const int r0   = (lane < 16) ? 0 : 8;
  v16h a[KT];
#pragma unroll
  for (int kt = 0; kt < KT; ++kt) a[kt] = load_a_frag(A, rb, lda, kt * 32);
  for (int nt = 0; nt < ntiles; ++nt) {
    if (nt + 1 < ntiles)
      __builtin_prefetch(B + (size_t)(nt + 1) * 512, 0, 3);
    v8f acc = {};
#pragma unroll
    for (int kt = 0; kt < KT; ++kt)
      acc = wmma16(a[kt], load_b_frag(B, kt * ntiles + nt), acc);
    const int col = nt * 16 + cr;
    float* cp = C + (size_t)(rb + r0) * ldc + col;
    if constexpr (HAS_ADD) {
      const float* ap = add + (size_t)(rb + r0) * ldadd + col;
      float av[8];
#pragma unroll
      for (int r = 0; r < 8; ++r) av[r] = ap[(size_t)r * ldadd];  // load clause
#pragma unroll
      for (int r = 0; r < 8; ++r) acc[r] += av[r];
    }
#pragma unroll
    for (int r = 0; r < 8; ++r) cp[(size_t)r * ldc] = acc[r];     // store clause
  }
}

// ---------------------------------------------------------------------------
// Edge geometry: r, cutoff, ea0, ea1, padded f16 Gaussian embedding (E x 32)
// ---------------------------------------------------------------------------
__global__ void edge_geom_kernel(const float* __restrict__ pos,
                                 const int* __restrict__ src,
                                 const int* __restrict__ dst,
                                 float* __restrict__ ea0, float* __restrict__ ea1,
                                 _Float16* __restrict__ embA, int E) {
  int e = blockIdx.x * blockDim.x + threadIdx.x;
  if (e >= E) return;
  int s = src[e], d = dst[e];
  float vx = pos[s * 3 + 0] - pos[d * 3 + 0];
  float vy = pos[s * 3 + 1] - pos[d * 3 + 1];
  float vz = pos[s * 3 + 2] - pos[d * 3 + 2];
  float r = sqrtf(vx * vx + vy * vy + vz * vz + 1e-12f);
  float inv = 1.f / r;
  // smooth_transition(r / 2): u = 2*(r/2 - 1) = r - 2
  float u = r - 2.f;
  float y = 0.5f * (1.f - __cosf(3.14159265358979f * u));
  y = (u > 0.f) ? 0.f : y;
  y = (u < -1.f) ? 1.f : y;
  ea0[e] = y;
  const float S3 = 1.7320508075688772f;
  float c = y * S3 * inv;
  ea1[e * 3 + 0] = c * vx;
  ea1[e * 3 + 1] = c * vy;
  ea1[e * 3 + 2] = c * vz;
  const float sq10 = 3.1622776601683795f;
#pragma unroll
  for (int i = 0; i < 10; ++i) {
    float t = (r - (2.f / 9.f) * (float)i) * (9.f / 2.f);
    embA[(size_t)e * 32 + i] = (_Float16)(__expf(-t * t) * sq10);
  }
#pragma unroll
  for (int i = 10; i < 32; ++i) embA[(size_t)e * 32 + i] = (_Float16)0.f;
}

// ---------------------------------------------------------------------------
// Fused edge conv: TDM-staged embedding tile -> radial MLP (2 WMMA GEMMs w/
// LDS bounce) -> tensor-product message -> atomic segment-sum scatter.
// One wave == 16 edges, 4 waves/block (block covers a contiguous 64-edge,
// 4KB embedding tile staged once via tensor_load_to_lds).
//   LAYER 0: x1=None : C2=32  (wA|wB),          D0=16
//   LAYER 1: full    : C2=160 (wA|wB|wC|wD),    D0=64, D1=16
//   LAYER 2: no mid1 : C2=80  (wA|wD),          D0=64, D1=16
// ---------------------------------------------------------------------------
template <int LAYER>
__global__ void __launch_bounds__(128)
edge_conv_kernel(const _Float16* __restrict__ embA, const float* __restrict__ ea0,
                 const float* __restrict__ ea1, const int* __restrict__ src,
                 const int* __restrict__ dst, const _Float16* __restrict__ B1,
                 const _Float16* __restrict__ B2, const float* __restrict__ y0,
                 const float* __restrict__ y1, float* __restrict__ s0,
                 float* __restrict__ s1, int E) {
  constexpr int C2   = (LAYER == 0) ? 32 : (LAYER == 1 ? 160 : 80);
  constexpr int C2NT = C2 / 16;
  constexpr int SD   = (LAYER == 0) ? 16 : 80;  // s0 row width
  __shared__ _Float16 estage[64][32];           // 4KB staged embedding tile
  __shared__ _Float16 hid[4][16][128];
  __shared__ float    wt[4][16][C2];
  const int wv    = threadIdx.x >> 5;
  const int lane  = threadIdx.x & 31;
  const int tile  = blockIdx.x * 4 + wv;
  const int ebase = tile * 16;

  // Async-stage this block's 64x32 f16 embedding tile into LDS (wave 0).
  if (threadIdx.x < 32)
    tdm_stage_1d(embA + (size_t)blockIdx.x * 2048, &estage[0][0], 2048u);
  __syncthreads();
  if (ebase >= E) return;

  // zero hidden pad columns 112..127 (K-tile 4 padding)
  for (int i = lane; i < 256; i += 32)
    hid[wv][i >> 4][112 + (i & 15)] = (_Float16)0.f;

  const int cr = lane & 15;
  const int r0 = (lane < 16) ? 0 : 8;

  // GEMM 1: hidden = relu(emb @ W1 / sqrt(10))  -> LDS (16 x 112 f16)
  v16h aemb = load_a_frag(&estage[0][0], wv * 16, 32, 0);
#pragma unroll
  for (int nt = 0; nt < 7; ++nt) {
    v8f acc = {};
    acc = wmma16(aemb, load_b_frag(B1, nt), acc);
#pragma unroll
    for (int r = 0; r < 8; ++r)
      hid[wv][r0 + r][nt * 16 + cr] = (_Float16)fmaxf(acc[r], 0.f);
  }

  // GEMM 2: W = hidden @ W2 / sqrt(100)  -> LDS (16 x C2 f32)
  const _Float16* hp = &hid[wv][0][0];
  v16h h0a = load_a_frag(hp, 0, 128, 0);
  v16h h1a = load_a_frag(hp, 0, 128, 32);
  v16h h2a = load_a_frag(hp, 0, 128, 64);
  v16h h3a = load_a_frag(hp, 0, 128, 96);
#pragma unroll
  for (int nt = 0; nt < C2NT; ++nt) {
    v8f acc = {};
    acc = wmma16(h0a, load_b_frag(B2, 0 * C2NT + nt), acc);
    acc = wmma16(h1a, load_b_frag(B2, 1 * C2NT + nt), acc);
    acc = wmma16(h2a, load_b_frag(B2, 2 * C2NT + nt), acc);
    acc = wmma16(h3a, load_b_frag(B2, 3 * C2NT + nt), acc);
#pragma unroll
    for (int r = 0; r < 8; ++r) wt[wv][r0 + r][nt * 16 + cr] = acc[r];
  }

  // Message + scatter: lane pairs (p, sub) split channels of edge ebase+p.
  const int p = lane >> 1, sub = lane & 1;
  const int e  = ebase + p;
  const int sn = src[e], dn = dst[e];
  const float a0  = ea0[e];
  const float a1x = ea1[e * 3 + 0], a1y = ea1[e * 3 + 1], a1z = ea1[e * 3 + 2];
  const float* wrow = &wt[wv][p][0];

  if constexpr (LAYER == 0) {
    for (int c = sub; c < 16; c += 2) {
      float g  = y0[(size_t)sn * 16 + c];
      float tB = wrow[16 + c] * g;
      atomicAdd(&s0[(size_t)dn * 16 + c], wrow[c] * g * a0);
      atomicAdd(&s1[((size_t)dn * 3 + 0) * 16 + c], tB * a1x);
      atomicAdd(&s1[((size_t)dn * 3 + 1) * 16 + c], tB * a1y);
      atomicAdd(&s1[((size_t)dn * 3 + 2) * 16 + c], tB * a1z);
    }
  } else {
    for (int c = sub; c < 64; c += 2) {
      float g = y0[(size_t)sn * 64 + c];
      atomicAdd(&s0[(size_t)dn * SD + c], wrow[c] * g * a0);
      if constexpr (LAYER == 1) {
        float tB = wrow[64 + c] * g;
        atomicAdd(&s1[((size_t)dn * 3 + 0) * SD + c], tB * a1x);
        atomicAdd(&s1[((size_t)dn * 3 + 1) * SD + c], tB * a1y);
        atomicAdd(&s1[((size_t)dn * 3 + 2) * SD + c], tB * a1z);
      }
    }
    constexpr int WD = (LAYER == 1) ? 144 : 64;
    const float RSQ3 = 0.5773502691896258f;
    for (int u = sub; u < 16; u += 2) {
      float gx = y1[((size_t)sn * 3 + 0) * 16 + u];
      float gy = y1[((size_t)sn * 3 + 1) * 16 + u];
      float gz = y1[((size_t)sn * 3 + 2) * 16 + u];
      float dotv = gx * a1x + gy * a1y + gz * a1z;
      atomicAdd(&s0[(size_t)dn * SD + 64 + u], wrow[WD + u] * dotv * RSQ3);
      if constexpr (LAYER == 1) {
        float wC = wrow[128 + u];
        atomicAdd(&s1[((size_t)dn * 3 + 0) * SD + 64 + u], wC * gx * a0);
        atomicAdd(&s1[((size_t)dn * 3 + 1) * SD + 64 + u], wC * gy * a0);
        atomicAdd(&s1[((size_t)dn * 3 + 2) * SD + 64 + u], wC * gz * a0);
      }
    }
  }
}

// ---------------------------------------------------------------------------
// Small elementwise kernels
// ---------------------------------------------------------------------------
__global__ void zero_f32(float* __restrict__ p, int n) {
  int i = blockIdx.x * blockDim.x + threadIdx.x;
  if (i < n) p[i] = 0.f;
}

__global__ void xz_to_h(const float* __restrict__ x, const float* __restrict__ z,
                        _Float16* __restrict__ xh, int N) {
  int i = blockIdx.x * blockDim.x + threadIdx.x;
  if (i >= N * 16) return;
  xh[i] = (_Float16)(x[i] * z[i >> 4]);
}

__global__ void cvt_pad_h(const float* __restrict__ s, _Float16* __restrict__ d,
                          int rows, int cs, int cd) {
  int i = blockIdx.x * blockDim.x + threadIdx.x;
  if (i >= rows * cd) return;
  int r = i / cd, c = i - r * cd;
  d[i] = (_Float16)((c < cs) ? s[(size_t)r * cs + c] : 0.f);
}

__global__ void gate_k(const float* __restrict__ h0raw, const float* __restrict__ h1raw,
                       const float* __restrict__ z, _Float16* __restrict__ h0h,
                       _Float16* __restrict__ h1h, int N) {
  int n = blockIdx.x * blockDim.x + threadIdx.x;
  if (n >= N) return;
  float zv = z[n];
  for (int c = 0; c < 64; ++c)
    h0h[(size_t)n * 64 + c] = (_Float16)(fmaxf(h0raw[(size_t)n * 80 + c], 0.f) * zv);
  for (int u = 0; u < 16; ++u) {
    float g = fmaxf(h0raw[(size_t)n * 80 + 64 + u], 0.f) * zv;
    for (int m = 0; m < 3; ++m)
      h1h[((size_t)n * 3 + m) * 16 + u] =
          (_Float16)(h1raw[((size_t)n * 3 + m) * 16 + u] * g);
  }
}

__global__ void final_reduce(const float* __restrict__ h0f, const int* __restrict__ batch,
                             float* __restrict__ out, int N) {
  int n = blockIdx.x * blockDim.x + threadIdx.x;
  if (n >= N) return;
  atomicAdd(&out[batch[n]], h0f[(size_t)n * 16] * 0.03125f);  // / sqrt(1024)
}

// ---------------------------------------------------------------------------
// Host orchestration
// ---------------------------------------------------------------------------
extern "C" void kernel_launch(void* const* d_in, const int* in_sizes, int n_in,
                              void* d_out, int out_size, void* d_ws, size_t ws_size,
                              hipStream_t stream) {
  const float* x        = (const float*)d_in[0];
  const float* z        = (const float*)d_in[1];
  const float* pos      = (const float*)d_in[2];
  const float* fc0_w1   = (const float*)d_in[3];
  const float* fc0_w2   = (const float*)d_in[4];
  const float* si0_w    = (const float*)d_in[5];
  const float* l0l1_w   = (const float*)d_in[6];
  const float* l0l2_w0  = (const float*)d_in[7];
  const float* l0l2_w1  = (const float*)d_in[8];
  const float* fc1_w1   = (const float*)d_in[9];
  const float* fc1_w2   = (const float*)d_in[10];
  const float* si1_w0   = (const float*)d_in[11];
  const float* si1_w1   = (const float*)d_in[12];
  const float* l1l1_w0  = (const float*)d_in[13];
  const float* l1l1_w1  = (const float*)d_in[14];
  const float* l1l2_w0  = (const float*)d_in[15];
  const float* l1l2_w1  = (const float*)d_in[16];
  const float* fcf_w1   = (const float*)d_in[17];
  const float* fcf_w2   = (const float*)d_in[18];
  const float* sif_w0   = (const float*)d_in[19];
  const float* lfl1_w0  = (const float*)d_in[20];
  const float* lfl1_w1  = (const float*)d_in[21];
  const float* lfl2_w0  = (const float*)d_in[22];
  const int* esrc  = (const int*)d_in[23];
  const int* edst  = (const int*)d_in[24];
  const int* batch = (const int*)d_in[25];

  const int N = in_sizes[0] / 16;
  const int E = in_sizes[23];
  auto cdiv = [](int a, int b) { return (a + b - 1) / b; };

  char* wsb = (char*)d_ws;
  size_t off = 0;
  auto alloc = [&](size_t bytes) -> void* {
    void* p = wsb + off;
    off = (off + bytes + 255) & ~(size_t)255;
    return p;
  };

  float*    ea0  = (float*)alloc((size_t)E * 4);
  float*    ea1  = (float*)alloc((size_t)E * 12);
  _Float16* embA = (_Float16*)alloc((size_t)E * 32 * 2);

  auto allocB = [&](int kt, int nt) { return (_Float16*)alloc((size_t)kt * nt * 512 * 2); };
  _Float16 *B1_0 = allocB(1, 7),  *B2_0 = allocB(4, 2);
  _Float16 *B1_1 = allocB(1, 7),  *B2_1 = allocB(4, 10);
  _Float16 *B1_f = allocB(1, 7),  *B2_f = allocB(4, 5);
  _Float16 *Bsi0 = allocB(1, 5),  *Bl0a = allocB(1, 1);
  _Float16 *Bl0c0 = allocB(1, 5), *Bl0c1 = allocB(1, 1);
  _Float16 *Bsi1a = allocB(2, 5), *Bsi1b = allocB(1, 1);
  _Float16 *Bl1a = allocB(2, 4),  *Bl1b = allocB(1, 1);
  _Float16 *Bl1c0 = allocB(3, 5), *Bl1c1 = allocB(3, 1);
  _Float16 *Bsif = allocB(2, 1),  *Blfa = allocB(2, 4);
  _Float16 *Blfb = allocB(1, 1),  *Blfc0 = allocB(3, 1);

  _Float16* xh     = (_Float16*)alloc((size_t)N * 16 * 2);
  float*    y0_0   = (float*)alloc((size_t)N * 16 * 4);
  float*    si0_0  = (float*)alloc((size_t)N * 80 * 4);
  float*    s0_0   = (float*)alloc((size_t)N * 16 * 4);
  float*    s1_0   = (float*)alloc((size_t)N * 48 * 4);
  _Float16* s0h_0  = (_Float16*)alloc((size_t)N * 16 * 2);
  _Float16* s1h_0  = (_Float16*)alloc((size_t)N * 48 * 2);
  float*    h0raw  = (float*)alloc((size_t)N * 80 * 4);
  float*    h1raw  = (float*)alloc((size_t)N * 48 * 4);
  _Float16* h0h    = (_Float16*)alloc((size_t)N * 64 * 2);
  _Float16* h1h    = (_Float16*)alloc((size_t)N * 48 * 2);
  float*    y0_1   = (float*)alloc((size_t)N * 64 * 4);
  float*    si0_1  = (float*)alloc((size_t)N * 80 * 4);
  float*    y1_1   = (float*)alloc((size_t)N * 48 * 4);
  float*    si1_1  = (float*)alloc((size_t)N * 48 * 4);
  float*    s0_1   = (float*)alloc((size_t)N * 80 * 4);
  float*    s1_1   = (float*)alloc((size_t)N * 240 * 4);
  _Float16* s0h_1  = (_Float16*)alloc((size_t)N * 96 * 2);
  _Float16* s1h_1  = (_Float16*)alloc((size_t)N * 288 * 2);
  float*    h0raw1 = (float*)alloc((size_t)N * 80 * 4);
  float*    h1raw1 = (float*)alloc((size_t)N * 48 * 4);
  float*    y0_f   = (float*)alloc((size_t)N * 64 * 4);
  float*    si0_f  = (float*)alloc((size_t)N * 16 * 4);
  float*    y1_f   = (float*)alloc((size_t)N * 48 * 4);
  float*    s0_f   = (float*)alloc((size_t)N * 80 * 4);
  _Float16* s0h_f  = (_Float16*)alloc((size_t)N * 96 * 2);
  float*    h0f    = (float*)alloc((size_t)N * 16 * 4);
  (void)alloc(256);  // guard for 32B A-fragment over-reads on last rows
  (void)ws_size; (void)n_in;

  // --- geometry + embedding ---
  edge_geom_kernel<<<cdiv(E, 256), 256, 0, stream>>>(pos, esrc, edst, ea0, ea1, embA, E);

  // --- pack all weights into WMMA B-fragments (scales folded in) ---
  auto pack = [&](const float* w, _Float16* dstp, int kt, int nt, int Kr, int Nr,
                  int ldw, int tr, float sc) {
    int total = kt * nt * 512;
    pack_b_kernel<<<cdiv(total, 256), 256, 0, stream>>>(w, dstp, kt, nt, Kr, Nr, ldw, tr, sc);
  };
  const float RS10 = 0.31622776601683794f;   // 1/sqrt(10)
  const float L2S16 = 0.03125f;              // 0.5 * (1/sqrt(16)) * (1/sqrt(16))
  const float L2S80 = 0.013975424859373686f; // 0.5 * 0.25 / sqrt(80)
  pack(fc0_w1, B1_0, 1, 7, 10, 100, 100, 0, RS10);
  pack(fc0_w2, B2_0, 4, 2, 100, 32, 32, 0, 0.1f);
  pack(fc1_w1, B1_1, 1, 7, 10, 100, 100, 0, RS10);
  pack(fc1_w2, B2_1, 4, 10, 100, 160, 160, 0, 0.1f);
  pack(fcf_w1, B1_f, 1, 7, 10, 100, 100, 0, RS10);
  pack(fcf_w2, B2_f, 4, 5, 100, 80, 80, 0, 0.1f);
  pack(si0_w,   Bsi0,  1, 5, 16, 80, 16, 1, 0.25f);
  pack(l0l1_w,  Bl0a,  1, 1, 16, 16, 16, 1, 0.25f);
  pack(l0l2_w0, Bl0c0, 1, 5, 16, 80, 16, 1, L2S16);
  pack(l0l2_w1, Bl0c1, 1, 1, 16, 16, 16, 1, L2S16);
  pack(si1_w0,  Bsi1a, 2, 5, 64, 80, 64, 1, 0.125f);
  pack(si1_w1,  Bsi1b, 1, 1, 16, 16, 16, 1, 0.25f);
  pack(l1l1_w0, Bl1a,  2, 4, 64, 64, 64, 1, 0.125f);
  pack(l1l1_w1, Bl1b,  1, 1, 16, 16, 16, 1, 0.25f);
  pack(l1l2_w0, Bl1c0, 3, 5, 80, 80, 80, 1, L2S80);
  pack(l1l2_w1, Bl1c1, 3, 1, 80, 16, 80, 1, L2S80);
  pack(sif_w0,  Bsif,  2, 1, 64, 1, 64, 1, 0.125f);
  pack(lfl1_w0, Blfa,  2, 4, 64, 64, 64, 1, 0.125f);
  pack(lfl1_w1, Blfb,  1, 1, 16, 16, 16, 1, 0.25f);
  pack(lfl2_w0, Blfc0, 3, 1, 80, 1, 80, 1, L2S80);

  auto gemm = [&](const _Float16* A, int lda, const _Float16* B, int rows, int kt,
                  int nt, float* C, int ldc, const float* add, int ldadd) {
    int rt = rows / 16;
    dim3 g(cdiv(rt, 8)), b(256);
    if (add) {
      if (kt == 1)
        gemm_wmma_kernel<1, true><<<g, b, 0, stream>>>(A, lda, B, rt, nt, C, ldc, add, ldadd);
      else if (kt == 2)
        gemm_wmma_kernel<2, true><<<g, b, 0, stream>>>(A, lda, B, rt, nt, C, ldc, add, ldadd);
      else
        gemm_wmma_kernel<3, true><<<g, b, 0, stream>>>(A, lda, B, rt, nt, C, ldc, add, ldadd);
    } else {
      if (kt == 1)
        gemm_wmma_kernel<1, false><<<g, b, 0, stream>>>(A, lda, B, rt, nt, C, ldc, nullptr, 0);
      else if (kt == 2)
        gemm_wmma_kernel<2, false><<<g, b, 0, stream>>>(A, lda, B, rt, nt, C, ldc, nullptr, 0);
      else
        gemm_wmma_kernel<3, false><<<g, b, 0, stream>>>(A, lda, B, rt, nt, C, ldc, nullptr, 0);
    }
  };
  const int egrid = cdiv(cdiv(E, 16), 4);

  // --- layer 0 ---
  xz_to_h<<<cdiv(N * 16, 256), 256, 0, stream>>>(x, z, xh, N);
  gemm(xh, 16, Bl0a, N, 1, 1, y0_0, 16, nullptr, 0);
  gemm(xh, 16, Bsi0, N, 1, 5, si0_0, 80, nullptr, 0);
  zero_f32<<<cdiv(N * 16, 256), 256, 0, stream>>>(s0_0, N * 16);
  zero_f32<<<cdiv(N * 48, 256), 256, 0, stream>>>(s1_0, N * 48);
  edge_conv_kernel<0><<<egrid, 128, 0, stream>>>(embA, ea0, ea1, esrc, edst, B1_0, B2_0,
                                                 y0_0, nullptr, s0_0, s1_0, E);
  cvt_pad_h<<<cdiv(N * 16, 256), 256, 0, stream>>>(s0_0, s0h_0, N, 16, 16);
  cvt_pad_h<<<cdiv(N * 48, 256), 256, 0, stream>>>(s1_0, s1h_0, 3 * N, 16, 16);
  gemm(s0h_0, 16, Bl0c0, N, 1, 5, h0raw, 80, si0_0, 80);
  gemm(s1h_0, 16, Bl0c1, 3 * N, 1, 1, h1raw, 16, nullptr, 0);
  gate_k<<<cdiv(N, 128), 128, 0, stream>>>(h0raw, h1raw, z, h0h, h1h, N);

  // --- layer 1 ---
  gemm(h0h, 64, Bl1a, N, 2, 4, y0_1, 64, nullptr, 0);
  gemm(h0h, 64, Bsi1a, N, 2, 5, si0_1, 80, nullptr, 0);
  gemm(h1h, 16, Bl1b, 3 * N, 1, 1, y1_1, 16, nullptr, 0);
  gemm(h1h, 16, Bsi1b, 3 * N, 1, 1, si1_1, 16, nullptr, 0);
  zero_f32<<<cdiv(N * 80, 256), 256, 0, stream>>>(s0_1, N * 80);
  zero_f32<<<cdiv(N * 240, 256), 256, 0, stream>>>(s1_1, N * 240);
  edge_conv_kernel<1><<<egrid, 128, 0, stream>>>(embA, ea0, ea1, esrc, edst, B1_1, B2_1,
                                                 y0_1, y1_1, s0_1, s1_1, E);
  cvt_pad_h<<<cdiv(N * 96, 256), 256, 0, stream>>>(s0_1, s0h_1, N, 80, 96);
  cvt_pad_h<<<cdiv(N * 288, 256), 256, 0, stream>>>(s1_1, s1h_1, 3 * N, 80, 96);
  gemm(s0h_1, 96, Bl1c0, N, 3, 5, h0raw1, 80, si0_1, 80);
  gemm(s1h_1, 96, Bl1c1, 3 * N, 3, 1, h1raw1, 16, si1_1, 16);
  gate_k<<<cdiv(N, 128), 128, 0, stream>>>(h0raw1, h1raw1, z, h0h, h1h, N);

  // --- final layer ---
  gemm(h0h, 64, Blfa, N, 2, 4, y0_f, 64, nullptr, 0);
  gemm(h0h, 64, Bsif, N, 2, 1, si0_f, 16, nullptr, 0);
  gemm(h1h, 16, Blfb, 3 * N, 1, 1, y1_f, 16, nullptr, 0);
  zero_f32<<<cdiv(N * 80, 256), 256, 0, stream>>>(s0_f, N * 80);
  edge_conv_kernel<2><<<egrid, 128, 0, stream>>>(embA, ea0, ea1, esrc, edst, B1_f, B2_f,
                                                 y0_f, y1_f, s0_f, nullptr, E);
  cvt_pad_h<<<cdiv(N * 96, 256), 256, 0, stream>>>(s0_f, s0h_f, N, 80, 96);
  gemm(s0h_f, 96, Blfc0, N, 3, 1, h0f, 16, si0_f, 16);

  // --- graph readout ---
  zero_f32<<<1, 32, 0, stream>>>((float*)d_out, out_size);
  final_reduce<<<cdiv(N, 256), 256, 0, stream>>>(h0f, batch, (float*)d_out, N);
}
